// LinearBlendSkinningModel_8409545966261
// MI455X (gfx1250) — matrix operand order repaired and checked by reading, not compiled, not run on hardware
//
#include <hip/hip_runtime.h>
#include <hip/hip_bf16.h>
#include <stdint.h>

// ---------------- problem constants ----------------
#define BB   1024          // batch
#define VV   6890          // vertices
#define PP   24            // joints
#define SDD  10            // shape dims
#define KK   217           // SD + 9*(P-1)
#define KP   224           // K padded to multiple of 32
#define NV3  20670         // 3*V
#define VTPB 64            // vertices per block (4 waves x 16)
#define NVT  108           // ceil(6890/64)
#define VPAD (NVT*VTPB)    // 6912 padded vertices
#define NPAD (VPAD*3)      // 20736 padded columns

typedef _Float16 v16h __attribute__((ext_vector_type(16)));
typedef _Float16 v8h  __attribute__((ext_vector_type(8)));
typedef float    v8f  __attribute__((ext_vector_type(8)));

// ---------------------------------------------------------------------------
// K0a: s_mat [217 x 20670] f32 -> ShT [NPAD x KP] f16, K-contiguous per column
// so each WMMA B-fragment lane reads 16 contiguous halfs (one 32B load).
// ---------------------------------------------------------------------------
__global__ void k_prep_smat(const float* __restrict__ s_mat,
                            _Float16* __restrict__ ShT) {
  int tid = blockIdx.x * blockDim.x + threadIdx.x;   // over KP * NPAD
  int k = tid / NPAD;
  int n = tid % NPAD;
  float v = (k < KK && n < NV3) ? s_mat[(size_t)k * NV3 + n] : 0.0f;
  ShT[(size_t)n * KP + k] = (_Float16)v;
}

// ---------------------------------------------------------------------------
// K0c: w_skin [6890 x 24] f32 -> Wh [VPAD x 32] f16 (K padded, rows padded 0)
// => WMMA A-fragment for the blend GEMM is two contiguous 16B loads per lane.
// ---------------------------------------------------------------------------
__global__ void k_prep_wskin(const float* __restrict__ w_skin,
                             _Float16* __restrict__ Wh) {
  int tid = blockIdx.x * blockDim.x + threadIdx.x;   // over VPAD*32
  int v = tid >> 5, p = tid & 31;
  float val = (v < VV && p < PP) ? w_skin[(size_t)v * PP + p] : 0.0f;
  Wh[(size_t)v * 32 + p] = (_Float16)val;
}

// ---------------------------------------------------------------------------
// K0b: JS[k][p][c] = sum_v j_reg[p,v]*s_mat[k,3v+c]  (k<10)
//      Jt[p][c]    = sum_v j_reg[p,v]*v_tmpl[3v+c]
// ---------------------------------------------------------------------------
__global__ void k_prep_joints(const float* __restrict__ s_mat,
                              const float* __restrict__ v_tmpl,
                              const float* __restrict__ j_reg,
                              float* __restrict__ JS,     // [10][24][3]
                              float* __restrict__ Jt) {   // [24][3]
  __shared__ float red[256];
  const int p = blockIdx.x, tid = threadIdx.x;
  float acc[33];
#pragma unroll
  for (int i = 0; i < 33; ++i) acc[i] = 0.0f;
  for (int v = tid; v < VV; v += 256) {
    float jr = j_reg[p * VV + v];
#pragma unroll
    for (int c = 0; c < 3; ++c) acc[30 + c] += jr * v_tmpl[3 * v + c];
#pragma unroll
    for (int k = 0; k < SDD; ++k)
#pragma unroll
      for (int c = 0; c < 3; ++c)
        acc[k * 3 + c] += jr * s_mat[(size_t)k * NV3 + 3 * v + c];
  }
#pragma unroll
  for (int j = 0; j < 33; ++j) {
    red[tid] = acc[j];
    __syncthreads();
    for (int s = 128; s > 0; s >>= 1) {
      if (tid < s) red[tid] += red[tid + s];
      __syncthreads();
    }
    if (tid == 0) {
      if (j < 30) JS[(j / 3) * (PP * 3) + p * 3 + (j % 3)] = red[0];
      else        Jt[p * 3 + (j - 30)] = red[0];
    }
    __syncthreads();
  }
}

// ---------------------------------------------------------------------------
// K1: per batch (1 wave): Rodrigues, J = Jt + x@JS, kinematic chain.
// Emits: feature row Fh[b][224] (f16) and per-batch transforms in WMMA
// B-fragment layout MhT[b][n=q(16)][k=joint(32)] f16, where
//   q = c*3+r (<9): rots[p][c][r],  q = 9+r: trans[p][r] - J[p].rots[p][:,r]
// ---------------------------------------------------------------------------
__global__ void k_batch(const float* __restrict__ x,
                        const float* __restrict__ JS,
                        const float* __restrict__ Jt,
                        _Float16* __restrict__ Fh,
                        _Float16* __restrict__ MhT) {
  __shared__ float xsh[SDD + 3 * (PP - 1)];
  __shared__ float Rsh[PP - 1][9];
  __shared__ float rot[PP][9];
  __shared__ float tr[PP][3];
  __shared__ float Jsh[PP][3];
  const int b = blockIdx.x, lane = threadIdx.x;

  for (int i = lane; i < SDD + 3 * (PP - 1); i += 32) xsh[i] = x[b * 79 + i];
  __syncthreads();

  if (lane < PP - 1) {                       // Rodrigues per joint
    float r0 = xsh[SDD + 3 * lane + 0];
    float r1 = xsh[SDD + 3 * lane + 1];
    float r2 = xsh[SDD + 3 * lane + 2];
    float th = sqrtf(r0 * r0 + r1 * r1 + r2 * r2);
    float inv = 1.0f / fmaxf(th, 1e-8f);
    float n0 = r0 * inv, n1 = r1 * inv, n2 = r2 * inv;
    float c = cosf(th), s = sinf(th), omc = 1.0f - c;
    float* R = Rsh[lane];
    R[0] = c + omc * n0 * n0;        R[1] = omc * n0 * n1 - s * n2;  R[2] = omc * n0 * n2 + s * n1;
    R[3] = omc * n1 * n0 + s * n2;   R[4] = c + omc * n1 * n1;       R[5] = omc * n1 * n2 - s * n0;
    R[6] = omc * n2 * n0 - s * n1;   R[7] = omc * n2 * n1 + s * n0;  R[8] = c + omc * n2 * n2;
  }
  if (lane < PP) {                           // J[b][p] = Jt + x@JS
#pragma unroll
    for (int c = 0; c < 3; ++c) {
      float a = Jt[lane * 3 + c];
#pragma unroll
      for (int k = 0; k < SDD; ++k) a += xsh[k] * JS[k * (PP * 3) + lane * 3 + c];
      Jsh[lane][c] = a;
    }
  }
  __syncthreads();

  if (lane == 0) {                           // sequential kinematic chain
#pragma unroll
    for (int q = 0; q < 9; ++q) rot[0][q] = (q == 0 || q == 4 || q == 8) ? 1.0f : 0.0f;
    tr[0][0] = Jsh[0][0]; tr[0][1] = Jsh[0][1]; tr[0][2] = Jsh[0][2];
    for (int i = 0; i < PP - 1; ++i) {       // node i+1, parent i
      const float* Ri = Rsh[i];
      const float* rp = rot[i];
      float rn[9];
#pragma unroll
      for (int a = 0; a < 3; ++a)
#pragma unroll
        for (int bb = 0; bb < 3; ++bb)       // rot_n = Ri^T @ rot_p
          rn[a * 3 + bb] = Ri[0 + a] * rp[0 + bb] + Ri[3 + a] * rp[3 + bb] + Ri[6 + a] * rp[6 + bb];
      float d0 = Jsh[i + 1][0] - Jsh[i][0];
      float d1 = Jsh[i + 1][1] - Jsh[i][1];
      float d2 = Jsh[i + 1][2] - Jsh[i][2];
#pragma unroll
      for (int q = 0; q < 9; ++q) rot[i + 1][q] = rn[q];
#pragma unroll
      for (int r = 0; r < 3; ++r)
        tr[i + 1][r] = d0 * rp[0 + r] + d1 * rp[3 + r] + d2 * rp[6 + r] + tr[i][r];
    }
  }
  __syncthreads();

  float ta[3] = {0.0f, 0.0f, 0.0f};          // t_adj for joint = lane
  if (lane < PP) {
#pragma unroll
    for (int r = 0; r < 3; ++r)
      ta[r] = tr[lane][r] - (Jsh[lane][0] * rot[lane][0 + r] +
                             Jsh[lane][1] * rot[lane][3 + r] +
                             Jsh[lane][2] * rot[lane][6 + r]);
  }
  // MhT[b][n][k]: lane = k (joint), iterate n. Coalesced half stores.
#pragma unroll
  for (int n = 0; n < 16; ++n) {
    float val = 0.0f;
    if (lane < PP) {
      if (n < 9)       val = rot[lane][n];
      else if (n < 12) val = ta[n - 9];
    }
    MhT[(size_t)b * 512 + n * 32 + lane] = (_Float16)val;
  }
  // feature row: [x_shape(10) | (R-I).flat(207) | zeros(7)]
  for (int idx = lane; idx < KP; idx += 32) {
    float val;
    if (idx < SDD) val = xsh[idx];
    else if (idx < KK) {
      int j = idx - SDD, i = j / 9, e = j % 9;
      val = Rsh[i][e] - ((e == 0 || e == 4 || e == 8) ? 1.0f : 0.0f);
    } else val = 0.0f;
    Fh[(size_t)b * KP + idx] = (_Float16)val;
  }
}

// ---------------------------------------------------------------------------
// K2: fused all-WMMA pipeline. block = 128 threads (4 waves).
// Wave tile: 16 batches x 16 vertices.
//  Phase 1: v_posed tile = F[16x224] @ ShT (3 n-tiles x 7 k-steps = 21 WMMA)
//  Phase 2: per batch m, T = w_skin[16x32] @ MhT[b][32x16] (16 WMMA),
//           then 6-FLOP/output epilogue, coalesced stores.
// ---------------------------------------------------------------------------
__global__ __launch_bounds__(128) void k_main(const _Float16* __restrict__ Fh,
                                              const _Float16* __restrict__ ShT,
                                              const float* __restrict__ v_tmpl,
                                              const _Float16* __restrict__ Wh,
                                              const _Float16* __restrict__ MhT,
                                              float* __restrict__ out) {
  __shared__ float vp[4][16][48];            // per-wave v_posed tile [m][c]
  __shared__ float Tl[4][16][16];            // per-wave blended transform [vl][q]
  const int tid = threadIdx.x;
  const int w = tid >> 5, lane = tid & 31;
  const int halfL = lane & 15, hi = lane >> 4;
  const int mb = blockIdx.x * 16;                    // batch tile base
  const int vbase = blockIdx.y * VTPB + w * 16;      // this wave's vertex base
  const int c0 = vbase * 3;                          // column base in 3V space

  // ---- Phase 1: v_posed GEMM ----
  v8f acc0 = {}, acc1 = {}, acc2 = {};
  const _Float16* Arow = Fh + (size_t)(mb + halfL) * KP + hi * 16;
  const _Float16* B0 = ShT + (size_t)(c0 + 0  + halfL) * KP + hi * 16;
  const _Float16* B1 = ShT + (size_t)(c0 + 16 + halfL) * KP + hi * 16;
  const _Float16* B2 = ShT + (size_t)(c0 + 32 + halfL) * KP + hi * 16;
#pragma unroll
  for (int ks = 0; ks < KP / 32; ++ks) {
    const int k0 = ks * 32;
    v8h alo = *(const v8h*)(Arow + k0);
    v8h ahi = *(const v8h*)(Arow + k0 + 8);
    v16h A = __builtin_shufflevector(alo, ahi, 0, 1, 2, 3, 4, 5, 6, 7,
                                               8, 9, 10, 11, 12, 13, 14, 15);
    v16h Bf0 = *(const v16h*)(B0 + k0);
    v16h Bf1 = *(const v16h*)(B1 + k0);
    v16h Bf2 = *(const v16h*)(B2 + k0);
    acc0 = __builtin_amdgcn_wmma_f32_16x16x32_f16(false, A, false, Bf0, (short)0, acc0, false, false);
    acc1 = __builtin_amdgcn_wmma_f32_16x16x32_f16(false, A, false, Bf1, (short)0, acc1, false, false);
    acc2 = __builtin_amdgcn_wmma_f32_16x16x32_f16(false, A, false, Bf2, (short)0, acc2, false, false);
  }

  // add v_tmpl, park tile in LDS (C layout: m = hi*8+i, n = halfL)
  int N0 = c0 + halfL, N1 = c0 + 16 + halfL, N2 = c0 + 32 + halfL;
  float vt0 = (N0 < NV3) ? v_tmpl[N0] : 0.0f;
  float vt1 = (N1 < NV3) ? v_tmpl[N1] : 0.0f;
  float vt2 = (N2 < NV3) ? v_tmpl[N2] : 0.0f;
#pragma unroll
  for (int i = 0; i < 8; ++i) {
    int m = hi * 8 + i;
    vp[w][m][0  + halfL] = acc0[i] + vt0;
    vp[w][m][16 + halfL] = acc1[i] + vt1;
    vp[w][m][32 + halfL] = acc2[i] + vt2;
  }
  __syncthreads();

  // ---- Phase 2: WMMA transform blend + tiny epilogue ----
  // A-fragment: skin weights for this wave's 16 vertices (reused for all 16 b)
  const _Float16* Wrow = Wh + (size_t)(vbase + halfL) * 32 + hi * 16;
  v8h wlo = *(const v8h*)(Wrow);
  v8h whi = *(const v8h*)(Wrow + 8);
  v16h Aw = __builtin_shufflevector(wlo, whi, 0, 1, 2, 3, 4, 5, 6, 7,
                                              8, 9, 10, 11, 12, 13, 14, 15);
  // per-lane output slots (consecutive => coalesced stores)
  const int oi0 = lane,      vl0 = oi0 / 3, r0_ = oi0 % 3;
  const int oi1 = lane + 32, vl1 = oi1 / 3, r1_ = oi1 % 3;
  const int va0 = vbase + vl0, va1 = vbase + vl1;

  for (int m = 0; m < 16; ++m) {
    const _Float16* Bp = MhT + (size_t)(mb + m) * 512 + (size_t)halfL * 32 + hi * 16;
    v16h Bm = *(const v16h*)Bp;
    v8f t = {};
    t = __builtin_amdgcn_wmma_f32_16x16x32_f16(false, Aw, false, Bm, (short)0, t, false, false);
#pragma unroll
    for (int i = 0; i < 8; ++i) Tl[w][hi * 8 + i][halfL] = t[i];
    // same-wave LDS ops are in-order; lanes read the wave's own T tile
    {
      float o = vp[w][m][vl0 * 3 + 0] * Tl[w][vl0][0 + r0_] +
                vp[w][m][vl0 * 3 + 1] * Tl[w][vl0][3 + r0_] +
                vp[w][m][vl0 * 3 + 2] * Tl[w][vl0][6 + r0_] +
                Tl[w][vl0][9 + r0_];
      if (va0 < VV)
        out[((size_t)(mb + m) * VV + vbase) * 3 + oi0] = o;
    }
    if (oi1 < 48) {
      float o = vp[w][m][vl1 * 3 + 0] * Tl[w][vl1][0 + r1_] +
                vp[w][m][vl1 * 3 + 1] * Tl[w][vl1][3 + r1_] +
                vp[w][m][vl1 * 3 + 2] * Tl[w][vl1][6 + r1_] +
                Tl[w][vl1][9 + r1_];
      if (va1 < VV)
        out[((size_t)(mb + m) * VV + vbase) * 3 + oi1] = o;
    }
  }
}

// ---------------------------------------------------------------------------
extern "C" void kernel_launch(void* const* d_in, const int* in_sizes, int n_in,
                              void* d_out, int out_size, void* d_ws, size_t ws_size,
                              hipStream_t stream) {
  const float* x      = (const float*)d_in[0];   // [1024,79]
  const float* s_mat  = (const float*)d_in[1];   // [217,20670]
  const float* v_tmpl = (const float*)d_in[2];   // [20670]
  const float* w_skin = (const float*)d_in[3];   // [6890,24]
  const float* j_reg  = (const float*)d_in[4];   // [24,6890]
  float* out = (float*)d_out;

  // workspace layout (bytes)
  char* ws = (char*)d_ws;
  _Float16* ShT = (_Float16*)(ws);                     // 20736*224*2 = 9,289,728
  _Float16* Fh  = (_Float16*)(ws + 9289728);           // 1024*224*2  =   458,752
  _Float16* Wh  = (_Float16*)(ws + 9748480);           // 6912*32*2   =   442,368
  _Float16* MhT = (_Float16*)(ws + 10190848);          // 1024*16*32*2= 1,048,576
  float*    JS  = (float*)(ws + 11239424);             // 720*4
  float*    Jt  = (float*)(ws + 11242304);             // 72*4

  // K0a: f16 transposed/padded copy of s_mat (B fragments for main GEMM)
  k_prep_smat<<<(KP * NPAD) / 256, 256, 0, stream>>>(s_mat, ShT);
  // K0c: f16 padded skin weights (A fragments for blend GEMM)
  k_prep_wskin<<<(VPAD * 32) / 256, 256, 0, stream>>>(w_skin, Wh);
  // K0b: joint regressor folded through blendshapes
  k_prep_joints<<<PP, 256, 0, stream>>>(s_mat, v_tmpl, j_reg, JS, Jt);
  // K1: per-batch rotations, chain, features, f16 transform fragments
  k_batch<<<BB, 32, 0, stream>>>(x, JS, Jt, Fh, MhT);
  // K2: fused WMMA GEMM + WMMA skinning blend
  dim3 grid(BB / 16, NVT);
  k_main<<<grid, 128, 0, stream>>>(Fh, ShT, v_tmpl, Wh, MhT, out);
}
